// ANIMASynthesis_86887188398451
// MI455X (gfx1250) — matrix operand order repaired and checked by reading, not compiled, not run on hardware
//
#include <hip/hip_runtime.h>
#include <hip/hip_bf16.h>

typedef __attribute__((ext_vector_type(16))) _Float16 v16h;
typedef __attribute__((ext_vector_type(8)))  _Float16 v8h;
typedef __attribute__((ext_vector_type(8)))  float    v8f;

#define T_LEN 1024
#define SDIM  8
#define DD    24
#define BDIM  8
#define OUTD  4
#define WAVES 4
#define TILE_M 16

// ---- packed B-fragment weight offsets in s_w (f16 elements) ----
// Every matrix is stored as NT blocks of 32 lanes x 16 contiguous halves
// (zero-padded): lane holds n = nt*16 + lane%16, k = (lane/16)*8 + (e/8)*16 + e%8.
#define PK_ENC 0        /* K=8  N=24 NT=2 -> 1024 */
#define PK_ZL  1024     /* K=24 N=24 NT=2 -> 1024 (x half of gru_z) */
#define PK_ZH  2048     /* I half */
#define PK_RL  3072
#define PK_RH  4096
#define PK_HL  5120
#define PK_HH  6144
#define PK_C   7168     /* K=24 N=8  NT=1 -> 512 */
#define PK_E   7680     /* K=8  N=24 NT=2 -> 1024 */
#define PK_PGL 8704     /* K=24 N=2  NT=1 -> 512 */
#define PK_PGH 9216
#define PK_WCL 9728
#define PK_WCH 10752
#define PK_IT  11776
#define PK_IC  12800
#define PK_IGL 13824
#define PK_IGH 14848
#define PK_AW  15872
#define PK_AI  16896
#define PK_AGL 17920
#define PK_AGH 18944
#define PK_PHI 19968    /* K=24 N=4 NT=1 -> 512 */
#define W_TOTAL 20480

// ---- zero-padded biases (32 f32 slots each) ----
#define B_ENC 0
#define B_Z   32
#define B_R   64
#define B_H   96
#define B_C   128
#define B_E   160
#define B_PG  192
#define B_WC  224
#define B_IG  256
#define B_AG  288
#define B_PHI 320
#define B_TOTAL 352

// ---- per-wave activation buffers, each [16][32] f16 (zero-padded tiles) ----
#define AB_O   0      /* obs (K=8) then compressed (K=8) */
#define AB_X   512    /* x */
#define AB_I   1024   /* I state (old -> new) */
#define AB_RI  1536   /* r*I */
#define AB_T   2048   /* temporal_out */
#define AB_C   2560   /* compressed_out */
#define AB_BL  3072   /* blended */
#define AB_W   3584   /* W_new */
#define AB_A   4096   /* A state */
#define AB_SZ  4608

#define LDSWAIT() asm volatile("s_wait_dscnt 0" ::: "memory")
#define LOG2E 1.4426950408889634f

struct PackedParams {
  const float *w_enc, *b_enc;
  const float *w_z,   *b_z;
  const float *w_r,   *b_r;
  const float *w_h,   *b_h;
  const float *w_c,   *b_c;
  const float *w_e,   *b_e;
  const float *w_pg,  *b_pg;
  const float *w_wc,  *b_wc;
  const float *w_it;
  const float *w_ic;
  const float *w_ig,  *b_ig;
  const float *w_aw;
  const float *w_ai;
  const float *w_ag,  *b_ag;
  const float *w_phi, *b_phi;
};

__device__ inline v8f zero8() {
  v8f z;
#pragma unroll
  for (int i = 0; i < 8; ++i) z[i] = 0.f;
  return z;
}
// branch-free tanh / sigmoid: single v_exp_f32 + v_rcp_f32 per element
__device__ inline v8f tanh8(v8f v) {
  v8f r;
#pragma unroll
  for (int i = 0; i < 8; ++i)
    r[i] = 1.f - 2.f * __builtin_amdgcn_rcpf(
                     1.f + __builtin_amdgcn_exp2f(2.f * LOG2E * v[i]));
  return r;
}
__device__ inline v8f sigm8(v8f v) {
  v8f r;
#pragma unroll
  for (int i = 0; i < 8; ++i)
    r[i] = __builtin_amdgcn_rcpf(
        1.f + __builtin_amdgcn_exp2f(-LOG2E * v[i]));
  return r;
}

// A-operand (16x32 f16) from a [16][32] LDS tile: two 16B vector loads.
__device__ inline v16h load_a32(const _Float16* act) {
  int lane = threadIdx.x & 31;
  const _Float16* p = act + (lane & 15) * 32 + (lane >> 4) * 8;
  v8h lo = *(const v8h*)p;
  v8h hh = *(const v8h*)(p + 16);
  v16h a;
#pragma unroll
  for (int e = 0; e < 8; ++e) { a[e] = lo[e]; a[8 + e] = hh[e]; }
  return a;
}
// B-operand from pre-packed fragment layout: 16 contiguous halves per lane.
__device__ inline v16h load_bp(const _Float16* base, int nt) {
  int lane = threadIdx.x & 31;
  const _Float16* p = base + ((nt * 32 + lane) << 4);
  v8h lo = *(const v8h*)p;
  v8h hh = *(const v8h*)(p + 8);
  v16h b;
#pragma unroll
  for (int e = 0; e < 8; ++e) { b[e] = lo[e]; b[8 + e] = hh[e]; }
  return b;
}
__device__ inline v8f wmma_f16(v16h a, v16h b, v8f c) {
  return __builtin_amdgcn_wmma_f32_16x16x32_f16(false, a, false, b, (short)0, c,
                                                false, false);
}
__device__ inline v8f gemm1(const _Float16* act, const _Float16* wpk, int nt,
                            v8f c) {
  return wmma_f16(load_a32(act), load_bp(wpk, nt), c);
}
__device__ inline v8f bias_add(v8f c, const float* b, int nb) {
  float bv = b[nb + ((int)threadIdx.x & 15)];
#pragma unroll
  for (int i = 0; i < 8; ++i) c[i] += bv;
  return c;
}
// unconditional D-fragment store into a [16][32] tile (16 columns)
__device__ inline void st32(v8f d, _Float16* buf, int nb) {
  int lane = threadIdx.x & 31;
  int n = nb + (lane & 15), hi = lane >> 4;
#pragma unroll
  for (int i = 0; i < 8; ++i) buf[(i + hi * 8) * 32 + n] = (_Float16)d[i];
}

// one-time pack: row-major [K][N] f32 -> padded B-fragment f16 layout
__device__ inline void pack_w(_Float16* dst, const float* src, int K, int N,
                              int NT, int tid, int nthr) {
  int total = NT * 512;
  for (int idx = tid; idx < total; idx += nthr) {
    int nt = idx >> 9;
    int lane = (idx >> 4) & 31;
    int e = idx & 15;
    int n = nt * 16 + (lane & 15);
    int k = (lane >> 4) * 8 + (e >> 3) * 16 + (e & 7);
    float v = (k < K && n < N) ? src[k * N + n] : 0.f;
    dst[idx] = (_Float16)v;
  }
}
__device__ inline void pack_b(float* dst, const float* src, int N, int tid,
                              int nthr) {
  for (int i = tid; i < 32; i += nthr) dst[i] = (i < N) ? src[i] : 0.f;
}

__global__ __launch_bounds__(32 * WAVES)
void anima_scan_kernel(const float* __restrict__ xg, float* __restrict__ outg,
                       PackedParams P) {
  __shared__ __align__(16) _Float16 s_w[W_TOTAL];
  __shared__ float s_bias[B_TOTAL];
  __shared__ __align__(16) _Float16 s_act[WAVES][AB_SZ];
  __shared__ float s_pw[WAVES][16 * 16];

  const int tid = threadIdx.x;
  const int nthr = blockDim.x;

  // ---- one-time weight packing (K=48 matrices split into two K=24 halves) --
  pack_w(s_w + PK_ENC, P.w_enc, SDIM, DD, 2, tid, nthr);
  pack_w(s_w + PK_ZL, P.w_z, DD, DD, 2, tid, nthr);
  pack_w(s_w + PK_ZH, P.w_z + DD * DD, DD, DD, 2, tid, nthr);
  pack_w(s_w + PK_RL, P.w_r, DD, DD, 2, tid, nthr);
  pack_w(s_w + PK_RH, P.w_r + DD * DD, DD, DD, 2, tid, nthr);
  pack_w(s_w + PK_HL, P.w_h, DD, DD, 2, tid, nthr);
  pack_w(s_w + PK_HH, P.w_h + DD * DD, DD, DD, 2, tid, nthr);
  pack_w(s_w + PK_C, P.w_c, DD, BDIM, 1, tid, nthr);
  pack_w(s_w + PK_E, P.w_e, BDIM, DD, 2, tid, nthr);
  pack_w(s_w + PK_PGL, P.w_pg, DD, 2, 1, tid, nthr);
  pack_w(s_w + PK_PGH, P.w_pg + DD * 2, DD, 2, 1, tid, nthr);
  pack_w(s_w + PK_WCL, P.w_wc, DD, DD, 2, tid, nthr);
  pack_w(s_w + PK_WCH, P.w_wc + DD * DD, DD, DD, 2, tid, nthr);
  pack_w(s_w + PK_IT, P.w_it, DD, DD, 2, tid, nthr);
  pack_w(s_w + PK_IC, P.w_ic, DD, DD, 2, tid, nthr);
  pack_w(s_w + PK_IGL, P.w_ig, DD, DD, 2, tid, nthr);
  pack_w(s_w + PK_IGH, P.w_ig + DD * DD, DD, DD, 2, tid, nthr);
  pack_w(s_w + PK_AW, P.w_aw, DD, DD, 2, tid, nthr);
  pack_w(s_w + PK_AI, P.w_ai, DD, DD, 2, tid, nthr);
  pack_w(s_w + PK_AGL, P.w_ag, DD, DD, 2, tid, nthr);
  pack_w(s_w + PK_AGH, P.w_ag + DD * DD, DD, DD, 2, tid, nthr);
  pack_w(s_w + PK_PHI, P.w_phi, DD, OUTD, 1, tid, nthr);
  pack_b(s_bias + B_ENC, P.b_enc, DD, tid, nthr);
  pack_b(s_bias + B_Z, P.b_z, DD, tid, nthr);
  pack_b(s_bias + B_R, P.b_r, DD, tid, nthr);
  pack_b(s_bias + B_H, P.b_h, DD, tid, nthr);
  pack_b(s_bias + B_C, P.b_c, BDIM, tid, nthr);
  pack_b(s_bias + B_E, P.b_e, DD, tid, nthr);
  pack_b(s_bias + B_PG, P.b_pg, 2, tid, nthr);
  pack_b(s_bias + B_WC, P.b_wc, DD, tid, nthr);
  pack_b(s_bias + B_IG, P.b_ig, DD, tid, nthr);
  pack_b(s_bias + B_AG, P.b_ag, DD, tid, nthr);
  pack_b(s_bias + B_PHI, P.b_phi, OUTD, tid, nthr);
  __syncthreads();

  const int wv = tid >> 5;
  const int lane = tid & 31;
  const int hi = lane >> 4;
  const int mbase = (blockIdx.x * WAVES + wv) * TILE_M;

  _Float16* bufO = &s_act[wv][AB_O];
  _Float16* bufX = &s_act[wv][AB_X];
  _Float16* bufI = &s_act[wv][AB_I];
  _Float16* bufRI = &s_act[wv][AB_RI];
  _Float16* bufT = &s_act[wv][AB_T];
  _Float16* bufC = &s_act[wv][AB_C];
  _Float16* bufBL = &s_act[wv][AB_BL];
  _Float16* bufW = &s_act[wv][AB_W];
  _Float16* bufAst = &s_act[wv][AB_A];
  float* pwb = s_pw[wv];

  v8f I0 = zero8(), I1 = zero8(), A0 = zero8(), A1 = zero8();
  st32(I0, bufI, 0);  st32(I1, bufI, 16);   // initial state copies
  st32(A0, bufAst, 0); st32(A1, bufAst, 16);
  LDSWAIT();

  for (int t = 0; t < T_LEN; ++t) {
    // ---- obs[16][8] -> f16 (2 lanes x float4 per row) ----
    {
      int row = lane >> 1, half = lane & 1;
      const float* sp =
          xg + ((size_t)(mbase + row) * T_LEN + (size_t)t) * SDIM + half * 4;
      float4 v = *(const float4*)sp;
      if (t + 1 < T_LEN) __builtin_prefetch(sp + SDIM, 0, 0);
      _Float16* dp = bufO + row * 32 + half * 4;
      dp[0] = (_Float16)v.x; dp[1] = (_Float16)v.y;
      dp[2] = (_Float16)v.z; dp[3] = (_Float16)v.w;
    }
    LDSWAIT();

    // ---- x = tanh(enc(obs)) ----
    v8f x0 = tanh8(bias_add(gemm1(bufO, s_w + PK_ENC, 0, zero8()), s_bias, B_ENC));
    v8f x1 = tanh8(bias_add(gemm1(bufO, s_w + PK_ENC, 1, zero8()), s_bias, B_ENC + 16));
    st32(x0, bufX, 0); st32(x1, bufX, 16);
    LDSWAIT();

    // ---- z, r, pathway logits: f([x|I]) = x@W_lo + I@W_hi ----
    v8f z0 = sigm8(bias_add(gemm1(bufI, s_w + PK_ZH, 0,
                                  gemm1(bufX, s_w + PK_ZL, 0, zero8())),
                            s_bias, B_Z));
    v8f z1 = sigm8(bias_add(gemm1(bufI, s_w + PK_ZH, 1,
                                  gemm1(bufX, s_w + PK_ZL, 1, zero8())),
                            s_bias, B_Z + 16));
    v8f r0 = sigm8(bias_add(gemm1(bufI, s_w + PK_RH, 0,
                                  gemm1(bufX, s_w + PK_RL, 0, zero8())),
                            s_bias, B_R));
    v8f r1 = sigm8(bias_add(gemm1(bufI, s_w + PK_RH, 1,
                                  gemm1(bufX, s_w + PK_RL, 1, zero8())),
                            s_bias, B_R + 16));
    v8f pwv = bias_add(gemm1(bufI, s_w + PK_PGH, 0,
                             gemm1(bufX, s_w + PK_PGL, 0, zero8())),
                       s_bias, B_PG);
    {
      int n = lane & 15;
#pragma unroll
      for (int i = 0; i < 8; ++i) pwb[(i + hi * 8) * 16 + n] = pwv[i];
    }
    st32(r0 * I0, bufRI, 0);
    st32(r1 * I1, bufRI, 16);
    LDSWAIT();

    // ---- h_cand, temporal_out ----
    v8f h0 = tanh8(bias_add(gemm1(bufRI, s_w + PK_HH, 0,
                                  gemm1(bufX, s_w + PK_HL, 0, zero8())),
                            s_bias, B_H));
    v8f h1 = tanh8(bias_add(gemm1(bufRI, s_w + PK_HH, 1,
                                  gemm1(bufX, s_w + PK_HL, 1, zero8())),
                            s_bias, B_H + 16));
    v8f t0 = (1.0f - z0) * I0 + z0 * h0;
    v8f t1 = (1.0f - z1) * I1 + z1 * h1;
    st32(t0, bufT, 0); st32(t1, bufT, 16);

    // ---- compress -> expand pathway ----
    v8f cp = tanh8(bias_add(gemm1(bufX, s_w + PK_C, 0, zero8()), s_bias, B_C));
    st32(cp, bufO, 0);
    LDSWAIT();
    v8f co0 = tanh8(bias_add(gemm1(bufO, s_w + PK_E, 0, zero8()), s_bias, B_E));
    v8f co1 = tanh8(bias_add(gemm1(bufO, s_w + PK_E, 1, zero8()), s_bias, B_E + 16));
    st32(co0, bufC, 0); st32(co1, bufC, 16);

    // ---- blended = softmax(pw) . [temporal, compressed] ----
    v8f bl0, bl1;
#pragma unroll
    for (int i = 0; i < 8; ++i) {
      int m = i + hi * 8;
      float p0 = __builtin_amdgcn_rcpf(
          1.f + __builtin_amdgcn_exp2f(LOG2E * (pwb[m * 16 + 1] - pwb[m * 16 + 0])));
      float p1 = 1.f - p0;
      bl0[i] = p0 * t0[i] + p1 * co0[i];
      bl1[i] = p0 * t1[i] + p1 * co1[i];
    }
    st32(bl0, bufBL, 0); st32(bl1, bufBL, 16);
    LDSWAIT();

    // ---- W_new = tanh(W_combine([x | A_old])) ----
    v8f w0 = tanh8(bias_add(gemm1(bufAst, s_w + PK_WCH, 0,
                                  gemm1(bufX, s_w + PK_WCL, 0, zero8())),
                            s_bias, B_WC));
    v8f w1 = tanh8(bias_add(gemm1(bufAst, s_w + PK_WCH, 1,
                                  gemm1(bufX, s_w + PK_WCL, 1, zero8())),
                            s_bias, B_WC + 16));
    st32(w0, bufW, 0); st32(w1, bufW, 16);
    LDSWAIT();

    // ---- I update ----
    v8f ic0 = gemm1(bufC, s_w + PK_IC, 0, gemm1(bufT, s_w + PK_IT, 0, zero8()));
    v8f ic1 = gemm1(bufC, s_w + PK_IC, 1, gemm1(bufT, s_w + PK_IT, 1, zero8()));
    v8f ig0 = sigm8(bias_add(gemm1(bufBL, s_w + PK_IGH, 0,
                                   gemm1(bufW, s_w + PK_IGL, 0, zero8())),
                             s_bias, B_IG));
    v8f ig1 = sigm8(bias_add(gemm1(bufBL, s_w + PK_IGH, 1,
                                   gemm1(bufW, s_w + PK_IGL, 1, zero8())),
                             s_bias, B_IG + 16));
    I0 = tanh8(ic0) * ig0 + (1.0f - ig0) * I0;
    I1 = tanh8(ic1) * ig1 + (1.0f - ig1) * I1;
    st32(I0, bufI, 0); st32(I1, bufI, 16);
    LDSWAIT();

    // ---- A update ----
    v8f ai0 = gemm1(bufI, s_w + PK_AI, 0, gemm1(bufW, s_w + PK_AW, 0, zero8()));
    v8f ai1 = gemm1(bufI, s_w + PK_AI, 1, gemm1(bufW, s_w + PK_AW, 1, zero8()));
    v8f ag0 = sigm8(bias_add(gemm1(bufI, s_w + PK_AGH, 0,
                                   gemm1(bufW, s_w + PK_AGL, 0, zero8())),
                             s_bias, B_AG));
    v8f ag1 = sigm8(bias_add(gemm1(bufI, s_w + PK_AGH, 1,
                                   gemm1(bufW, s_w + PK_AGL, 1, zero8())),
                             s_bias, B_AG + 16));
    v8f ta0 = tanh8(ai0), ta1 = tanh8(ai1);
    A0 = ta0 * ag0 + 0.1f * ta0;
    A1 = ta1 * ag1 + 0.1f * ta1;
    st32(A0, bufAst, 0); st32(A1, bufAst, 16);
    LDSWAIT();

    // ---- action = phi(A_new) + b ----
    v8f act = bias_add(gemm1(bufAst, s_w + PK_PHI, 0, zero8()), s_bias, B_PHI);
    {
      int n = lane & 15;
      if (n < OUTD) {
#pragma unroll
        for (int i = 0; i < 8; ++i) {
          int m = i + hi * 8;
          outg[((size_t)(mbase + m) * T_LEN + (size_t)t) * OUTD + n] = act[i];
        }
      }
    }
  }
}

extern "C" void kernel_launch(void* const* d_in, const int* in_sizes, int n_in,
                              void* d_out, int out_size, void* d_ws,
                              size_t ws_size, hipStream_t stream) {
  (void)in_sizes; (void)n_in; (void)d_ws; (void)ws_size; (void)out_size;
  const float* xg = (const float*)d_in[0];
  PackedParams P;
  P.w_enc = (const float*)d_in[1];  P.b_enc = (const float*)d_in[2];
  P.w_z   = (const float*)d_in[3];  P.b_z   = (const float*)d_in[4];
  P.w_r   = (const float*)d_in[5];  P.b_r   = (const float*)d_in[6];
  P.w_h   = (const float*)d_in[7];  P.b_h   = (const float*)d_in[8];
  P.w_c   = (const float*)d_in[9];  P.b_c   = (const float*)d_in[10];
  P.w_e   = (const float*)d_in[11]; P.b_e   = (const float*)d_in[12];
  P.w_pg  = (const float*)d_in[13]; P.b_pg  = (const float*)d_in[14];
  P.w_wc  = (const float*)d_in[15]; P.b_wc  = (const float*)d_in[16];
  P.w_it  = (const float*)d_in[17];
  P.w_ic  = (const float*)d_in[18];
  P.w_ig  = (const float*)d_in[19]; P.b_ig  = (const float*)d_in[20];
  P.w_aw  = (const float*)d_in[21];
  P.w_ai  = (const float*)d_in[22];
  P.w_ag  = (const float*)d_in[23]; P.b_ag  = (const float*)d_in[24];
  P.w_phi = (const float*)d_in[25]; P.b_phi = (const float*)d_in[26];

  const int batch = 4096;
  dim3 grid(batch / (TILE_M * WAVES));
  dim3 block(32 * WAVES);
  anima_scan_kernel<<<grid, block, 0, stream>>>(xg, (float*)d_out, P);
}
